// NSA_Fusion_55542517072409
// MI455X (gfx1250) — matrix operand order repaired and checked by reading, not compiled, not run on hardware
//
#include <hip/hip_runtime.h>
#include <hip/hip_bf16.h>
#include <stdint.h>

// ---------------------------------------------------------------------------
// NSA transformer forward for MI455X (gfx1250, wave32, WMMA)
// ---------------------------------------------------------------------------

typedef __bf16 bf16_t;
typedef __attribute__((ext_vector_type(16))) __bf16 v16bf;
typedef __attribute__((ext_vector_type(8)))  __bf16 v8bf;
typedef __attribute__((ext_vector_type(8)))  float  v8f;
typedef __attribute__((ext_vector_type(4)))  int    v4i;

#define NH    8
#define HD    64
#define DM    512
#define DFF   1024
#define SEQ   2048
#define BAT   2
#define TOK   (BAT*SEQ)
#define NBLK  512          /* nb = SEQ/BC */
#define NCMP  513          /* nb + 1 mem token */
#define SCALE 0.125f       /* 1/sqrt(64) */
#define NEGB  (-1e9f)

// ---- CDNA5 async global->LDS staging (guarded: falls back to reg staging) --
#if defined(__has_builtin)
#  if __has_builtin(__builtin_amdgcn_global_load_async_to_lds_b128) && \
      __has_builtin(__builtin_amdgcn_s_wait_asynccnt)
#    define USE_ASYNC_LDS 1
#  endif
#endif
#ifndef USE_ASYNC_LDS
#  define USE_ASYNC_LDS 0
#endif

#if USE_ASYNC_LDS
typedef __attribute__((address_space(1))) v4i* gv4ip_t;   // global int4*
typedef __attribute__((address_space(3))) v4i* lv4ip_t;   // LDS int4*
__device__ __forceinline__ void async_cp16(const void* g, void* l) {
  __builtin_amdgcn_global_load_async_to_lds_b128(
      (gv4ip_t)(v4i*)(uintptr_t)g, (lv4ip_t)(v4i*)l, 0, 0);
}
#endif

__device__ __forceinline__ float bf2f(bf16_t x) {
  unsigned short u = __builtin_bit_cast(unsigned short, x);
  unsigned v = ((unsigned)u) << 16;
  return __builtin_bit_cast(float, v);
}
__device__ __forceinline__ bf16_t f2bf(float f) {
  unsigned u = __builtin_bit_cast(unsigned, f);
  unsigned r = u + 0x7FFFu + ((u >> 16) & 1u);   // round-to-nearest-even
  unsigned short s = (unsigned short)(r >> 16);
  return __builtin_bit_cast(bf16_t, s);
}
__device__ __forceinline__ float wred_sum(float v) {
  #pragma unroll
  for (int o = 16; o; o >>= 1) v += __shfl_xor(v, o, 32);
  return v;
}
__device__ __forceinline__ float wred_max(float v) {
  #pragma unroll
  for (int o = 16; o; o >>= 1) v = fmaxf(v, __shfl_xor(v, o, 32));
  return v;
}

// ---------------------------------------------------------------------------
// fp32 -> bf16 conversion (weights / staging)
// ---------------------------------------------------------------------------
__global__ void cvt_f32_bf16(const float* __restrict__ s, bf16_t* __restrict__ d, int n) {
  int i = blockIdx.x * blockDim.x + threadIdx.x;
  if (i < n) d[i] = f2bf(s[i]);
}

// ---------------------------------------------------------------------------
// LayerNorm (one wave per row of D=512), bf16 output
// ---------------------------------------------------------------------------
__global__ __launch_bounds__(256) void ln_bf16(
    const float* __restrict__ x, const float* __restrict__ g,
    const float* __restrict__ b, bf16_t* __restrict__ h, int T) {
  int wave = threadIdx.x >> 5, lane = threadIdx.x & 31;
  int row = blockIdx.x * 8 + wave;
  if (row >= T) return;
  const float* xr = x + (size_t)row * DM;
  float s = 0.f, ss = 0.f;
  #pragma unroll
  for (int i = 0; i < DM / 32; i++) { float v = xr[lane + 32 * i]; s += v; ss += v * v; }
  s = wred_sum(s); ss = wred_sum(ss);
  float mean = s * (1.0f / DM);
  float var  = ss * (1.0f / DM) - mean * mean;
  float rstd = rsqrtf(var + 1e-5f);
  bf16_t* hr = h + (size_t)row * DM;
  #pragma unroll
  for (int i = 0; i < DM / 32; i++) {
    int c = lane + 32 * i;
    hr[c] = f2bf((xr[c] - mean) * rstd * g[c] + b[c]);
  }
}

// ---------------------------------------------------------------------------
// bf16 WMMA GEMM: C[M,N] = A[M,K] @ B[K,N]  (fp32 accumulate)
//   mode 0: Cf = acc (+bias)
//   mode 1: Cf = resid + acc (+bias)              (residual; resid may == Cf)
//   mode 2: Cbf = bf16(leaky_relu(acc + bias))    (FF1)
// Block tile 128x128, 8 waves (4 along M x 2 along N), each wave 32x64 via
// 2x4 v_wmma_f32_16x16x32_bf16 tiles. K step 32, double-buffered LDS,
// one barrier per K step. A tile staged via async global->LDS when available.
// ---------------------------------------------------------------------------
#define BM 128
#define BN 128
#define BK 32
#define LDK (BK + 8)

__global__ __launch_bounds__(256) void gemm_wmma(
    const bf16_t* __restrict__ A, const bf16_t* __restrict__ Bw,
    float* __restrict__ Cf, bf16_t* __restrict__ Cbf,
    const float* __restrict__ bias, const float* __restrict__ resid,
    int M, int N, int K, int mode) {
  __shared__ __align__(16) bf16_t As[2][BM][LDK];
  __shared__ __align__(16) bf16_t Bs[2][BN][LDK];   // transposed: Bs[n][k]

  int tid  = threadIdx.x;
  int lane = tid & 31, wave = tid >> 5;
  int m0 = blockIdx.y * BM, n0 = blockIdx.x * BN;
  int waveM = wave & 3, waveN = wave >> 2;          // 4 x 2 wave grid
  int mw = waveM * 32, nw = waveN * 64;

  v8f acc[2][4];
  {
    v8f z;
    #pragma unroll
    for (int e = 0; e < 8; e++) z[e] = 0.f;
    #pragma unroll
    for (int i = 0; i < 2; i++)
      #pragma unroll
      for (int j = 0; j < 4; j++) acc[i][j] = z;
  }

  int arow = tid >> 1, acol = (tid & 1) * 16;       // A: 128 rows x 32 cols
  int brow = tid >> 3, bcol = (tid & 7) * 16;       // B: 32 rows x 128 cols

  uint4 rB0, rB1;                                    // B staging registers
#if !USE_ASYNC_LDS
  uint4 rA0, rA1;                                    // A staging registers
#endif

  const int KT = K / BK;

  // ---- prologue: stage tile 0 into buffer 0 ----
  {
    const bf16_t* ga = A + (size_t)(m0 + arow) * K + acol;
#if USE_ASYNC_LDS
    async_cp16(ga,     &As[0][arow][acol]);
    async_cp16(ga + 8, &As[0][arow][acol + 8]);
#else
    rA0 = *(const uint4*)(ga);
    rA1 = *(const uint4*)(ga + 8);
    *(uint4*)(&As[0][arow][acol])     = rA0;
    *(uint4*)(&As[0][arow][acol + 8]) = rA1;
#endif
    const bf16_t* gb = Bw + (size_t)brow * N + n0 + bcol;
    rB0 = *(const uint4*)(gb);
    rB1 = *(const uint4*)(gb + 8);
    const bf16_t* h0 = (const bf16_t*)&rB0;
    const bf16_t* h1 = (const bf16_t*)&rB1;
    #pragma unroll
    for (int i = 0; i < 8; i++) Bs[0][bcol + i][brow]     = h0[i];
    #pragma unroll
    for (int i = 0; i < 8; i++) Bs[0][bcol + 8 + i][brow] = h1[i];
  }
#if USE_ASYNC_LDS
  __builtin_amdgcn_s_wait_asynccnt(0);
#endif
  __syncthreads();

  for (int kt = 0; kt < KT; ++kt) {
    int cur = kt & 1, nxt = cur ^ 1;

    // issue next-tile global traffic before this tile's WMMAs
    if (kt + 1 < KT) {
      int k1 = (kt + 1) * BK;
      const bf16_t* ga = A + (size_t)(m0 + arow) * K + k1 + acol;
#if USE_ASYNC_LDS
      async_cp16(ga,     &As[nxt][arow][acol]);
      async_cp16(ga + 8, &As[nxt][arow][acol + 8]);
#else
      rA0 = *(const uint4*)(ga);
      rA1 = *(const uint4*)(ga + 8);
#endif
      const bf16_t* gb = Bw + (size_t)(k1 + brow) * N + n0 + bcol;
      rB0 = *(const uint4*)(gb);
      rB1 = *(const uint4*)(gb + 8);
      if (kt + 2 < KT) {
        __builtin_prefetch(A + (size_t)(m0 + arow) * K + (kt + 2) * BK + acol, 0, 0);
        __builtin_prefetch(Bw + (size_t)((kt + 2) * BK + brow) * N + n0 + bcol, 0, 0);
      }
    }

    // --- fragments per CDNA5 WMMA VGPR striping ---
    int r   = lane & 15;
    int kbA = (lane < 16) ? 0 : 8;    // A: lanes 0-15 K{0..7,16..23}; 16-31 K{8..15,24..31}
    int kbB = (lane < 16) ? 0 : 8;    // B: lanes 0-15 K{0..7,16..23}; 16-31 K{8..15,24..31}

    v16bf af[2], bfm[4];
    #pragma unroll
    for (int mi = 0; mi < 2; mi++) {
      const bf16_t* p = &As[cur][mw + 16 * mi + r][kbA];
      v8bf a0 = *(const v8bf*)(p);
      v8bf a1 = *(const v8bf*)(p + 16);
      #pragma unroll
      for (int i = 0; i < 8; i++) { af[mi][i] = a0[i]; af[mi][i + 8] = a1[i]; }
    }
    #pragma unroll
    for (int ni = 0; ni < 4; ni++) {
      const bf16_t* p = &Bs[cur][nw + 16 * ni + r][kbB];
      v8bf b0 = *(const v8bf*)(p);
      v8bf b1 = *(const v8bf*)(p + 16);
      #pragma unroll
      for (int i = 0; i < 8; i++) { bfm[ni][i] = b0[i]; bfm[ni][i + 8] = b1[i]; }
    }
    #pragma unroll
    for (int mi = 0; mi < 2; mi++)
      #pragma unroll
      for (int ni = 0; ni < 4; ni++)
        acc[mi][ni] = __builtin_amdgcn_wmma_f32_16x16x32_bf16(
            false, af[mi], false, bfm[ni], (short)0, acc[mi][ni], false, false);

    // store next tile into the other buffer (waits land after WMMAs)
    if (kt + 1 < KT) {
#if !USE_ASYNC_LDS
      *(uint4*)(&As[nxt][arow][acol])     = rA0;
      *(uint4*)(&As[nxt][arow][acol + 8]) = rA1;
#endif
      const bf16_t* h0 = (const bf16_t*)&rB0;
      const bf16_t* h1 = (const bf16_t*)&rB1;
      #pragma unroll
      for (int i = 0; i < 8; i++) Bs[nxt][bcol + i][brow]     = h0[i];
      #pragma unroll
      for (int i = 0; i < 8; i++) Bs[nxt][bcol + 8 + i][brow] = h1[i];
    }
#if USE_ASYNC_LDS
    __builtin_amdgcn_s_wait_asynccnt(0);
#endif
    __syncthreads();
  }

  // epilogue: VGPR rr -> M = rr + 8*(lane>=16); column = lane&15
  int cbase = n0 + nw + (lane & 15);
  int rbase = m0 + mw + ((lane >> 4) << 3);
  #pragma unroll
  for (int mi = 0; mi < 2; mi++) {
    #pragma unroll
    for (int ni = 0; ni < 4; ni++) {
      int col = cbase + 16 * ni;
      float bv = bias ? bias[col] : 0.f;
      #pragma unroll
      for (int rr = 0; rr < 8; rr++) {
        int row = rbase + 16 * mi + rr;
        float v = acc[mi][ni][rr] + bv;
        size_t idx = (size_t)row * N + col;
        if (mode == 0)      Cf[idx] = v;
        else if (mode == 1) Cf[idx] = resid[idx] + v;
        else {
          v = (v > 0.f) ? v : 0.01f * v;   // LeakyReLU(0.01)
          Cbf[idx] = f2bf(v);
        }
      }
    }
  }
}

// ---------------------------------------------------------------------------
// Compressed K/V: out[b][0] = mem token; out[b][m>=1] = mean of block m-1
// ---------------------------------------------------------------------------
__global__ void compress_kv(const float* __restrict__ kv, const float* __restrict__ mem,
                            float* __restrict__ out) {
  int idx = blockIdx.x * blockDim.x + threadIdx.x;
  int total = BAT * NCMP * DM;
  if (idx >= total) return;
  int col = idx % DM;
  int m   = (idx / DM) % NCMP;
  int b   = idx / (DM * NCMP);
  float v;
  if (m == 0) v = mem[col];
  else {
    int base = b * SEQ + (m - 1) * 4;
    v = 0.25f * (kv[(size_t)(base + 0) * DM + col] + kv[(size_t)(base + 1) * DM + col] +
                 kv[(size_t)(base + 2) * DM + col] + kv[(size_t)(base + 3) * DM + col]);
  }
  out[idx] = v;
}

// ---------------------------------------------------------------------------
// Fused NSA attention: one block per (b,n), one wave per head.
// ---------------------------------------------------------------------------
__global__ __launch_bounds__(256) void nsa_attn(
    const float* __restrict__ q, const float* __restrict__ k, const float* __restrict__ v,
    const float* __restrict__ kc, const float* __restrict__ vc,
    const bf16_t* __restrict__ h, const float* __restrict__ Wg,
    bf16_t* __restrict__ o) {
  __shared__ float sc_s[NH][520];
  __shared__ float imp_s[NBLK];
  __shared__ float qh_s[NH][HD];
  __shared__ int   sel_s[2];

  int tok = blockIdx.x;
  int b = tok / SEQ, n = tok % SEQ;
  int hh = threadIdx.x >> 5, lane = threadIdx.x & 31;

  for (int i = threadIdx.x; i < NBLK; i += 256) imp_s[i] = 0.f;
  const float* qp = q + (size_t)tok * DM + hh * HD;
  qh_s[hh][lane]      = qp[lane];
  qh_s[hh][lane + 32] = qp[lane + 32];
  __syncthreads();

  int nvis = (n + 1) >> 2;   // visible pooled blocks (block_end <= n)
  int cnt  = nvis + 1;       // + mem token (m==0)

  // pass 1: scores over visible compressed keys
  float lmax = -3.0e38f;
  for (int m = lane; m < cnt; m += 32) {
    const float* kp = kc + ((size_t)(b * NCMP + m)) * DM + hh * HD;
    float s = 0.f;
    #pragma unroll 8
    for (int d = 0; d < HD; d++) s += qh_s[hh][d] * kp[d];
    s *= SCALE;
    sc_s[hh][m] = s;
    lmax = fmaxf(lmax, s);
  }
  float rmax = wred_max(lmax);
  float lsum = 0.f;
  for (int m = lane; m < cnt; m += 32) {
    float p = __expf(sc_s[hh][m] - rmax);
    sc_s[hh][m] = p;
    lsum += p;
  }
  float rsum = wred_sum(lsum);
  float inv  = 1.0f / rsum;

  // head-averaged importance of real blocks (pc[...,1:].mean over heads)
  for (int m = lane; m < cnt; m += 32)
    if (m >= 1) atomicAdd(&imp_s[m - 1], sc_s[hh][m] * inv * (1.0f / NH));

  // pass 2: out_c for lane-owned dims
  float oc0 = 0.f, oc1 = 0.f;
  for (int m = 0; m < cnt; m++) {
    float p = sc_s[hh][m] * inv;
    const float* vp = vc + ((size_t)(b * NCMP + m)) * DM + hh * HD;
    oc0 += p * vp[lane];
    oc1 += p * vp[lane + 32];
  }
  __syncthreads();

  // top-2 block selection (wave 0); ties -> lower index (matches lax.top_k)
  if (hh == 0) {
    float bv = -3.0e38f; int bi = 1 << 30;
    for (int i = lane; i < NBLK; i += 32) {
      float vv = imp_s[i];
      if (vv > bv || (vv == bv && i < bi)) { bv = vv; bi = i; }
    }
    #pragma unroll
    for (int off = 16; off; off >>= 1) {
      float ov = __shfl_xor(bv, off, 32);
      int   oi = __shfl_xor(bi, off, 32);
      if (ov > bv || (ov == bv && oi < bi)) { bv = ov; bi = oi; }
    }
    int s0 = bi;
    bv = -3.0e38f; bi = 1 << 30;
    for (int i = lane; i < NBLK; i += 32) {
      if (i == s0) continue;
      float vv = imp_s[i];
      if (vv > bv || (vv == bv && i < bi)) { bv = vv; bi = i; }
    }
    #pragma unroll
    for (int off = 16; off; off >>= 1) {
      float ov = __shfl_xor(bv, off, 32);
      int   oi = __shfl_xor(bi, off, 32);
      if (ov > bv || (ov == bv && oi < bi)) { bv = ov; bi = oi; }
    }
    if (lane == 0) { sel_s[0] = s0; sel_s[1] = bi; }
  }
  __syncthreads();
  int s0 = sel_s[0], s1 = sel_s[1];

  // fine selection branch: 8 candidate positions, lanes 0..7 score one each
  int pos = 0; float ssel = NEGB;
  if (lane < 8) {
    int blk = (lane < 4) ? s0 : s1;
    pos = blk * 4 + (lane & 3);
    if (pos <= n) {
      const float* kp = k + ((size_t)(b * SEQ + pos)) * DM + hh * HD;
      float s = 0.f;
      #pragma unroll 8
      for (int d = 0; d < HD; d++) s += qh_s[hh][d] * kp[d];
      ssel = s * SCALE;
    }
  } else ssel = -3.0e38f;
  float smax = wred_max(ssel);
  float e    = (lane < 8) ? __expf(ssel - smax) : 0.f;
  float esum = wred_sum(e);
  float psel = e / esum;
  float os0 = 0.f, os1 = 0.f;
  #pragma unroll
  for (int j = 0; j < 8; j++) {
    float pj  = __shfl(psel, j, 32);
    int   pjp = __shfl(pos,  j, 32);
    const float* vp = v + ((size_t)(b * SEQ + pjp)) * DM + hh * HD;
    os0 += pj * vp[lane];
    os1 += pj * vp[lane + 32];
  }

  // sliding window (self + previous token)
  int prev = (n > 0) ? n - 1 : 0;
  const float* kp0 = k + (size_t)tok * DM + hh * HD;
  const float* kp1 = k + ((size_t)(b * SEQ + prev)) * DM + hh * HD;
  float p0 = qh_s[hh][lane] * kp0[lane] + qh_s[hh][lane + 32] * kp0[lane + 32];
  float p1 = qh_s[hh][lane] * kp1[lane] + qh_s[hh][lane + 32] * kp1[lane + 32];
  float sw0 = wred_sum(p0) * SCALE;
  float sw1 = wred_sum(p1) * SCALE;
  if (n == 0) sw1 = NEGB;
  float wm = fmaxf(sw0, sw1);
  float e0 = __expf(sw0 - wm), e1 = __expf(sw1 - wm);
  float wden = 1.0f / (e0 + e1);
  float pw0 = e0 * wden, pw1 = e1 * wden;
  const float* vp0 = v + (size_t)tok * DM + hh * HD;
  const float* vp1 = v + ((size_t)(b * SEQ + prev)) * DM + hh * HD;
  float ow0 = pw0 * vp0[lane]      + pw1 * vp1[lane];
  float ow1 = pw0 * vp0[lane + 32] + pw1 * vp1[lane + 32];

  // gates = sigmoid(h @ Wg) per (head, branch)
  const bf16_t* hr = h + (size_t)tok * DM;
  float gg[3];
  #pragma unroll
  for (int j = 0; j < 3; j++) {
    float part = 0.f;
    for (int i = lane; i < DM; i += 32)
      part += bf2f(hr[i]) * Wg[(size_t)i * (3 * NH) + hh * 3 + j];
    part = wred_sum(part);
    gg[j] = 1.0f / (1.0f + __expf(-part));
  }

  float o0 = gg[0] * oc0 + gg[1] * os0 + gg[2] * ow0;
  float o1 = gg[0] * oc1 + gg[1] * os1 + gg[2] * ow1;
  bf16_t* orow = o + (size_t)tok * DM + hh * HD;
  orow[lane]      = f2bf(o0);
  orow[lane + 32] = f2bf(o1);
}

// ---------------------------------------------------------------------------
// Host orchestration
// ---------------------------------------------------------------------------
extern "C" void kernel_launch(void* const* d_in, const int* in_sizes, int n_in,
                              void* d_out, int out_size, void* d_ws, size_t ws_size,
                              hipStream_t stream) {
  (void)in_sizes; (void)n_in; (void)out_size; (void)ws_size;

  const float* x    = (const float*)d_in[0];
  const float* lng  = (const float*)d_in[1];
  const float* lnb  = (const float*)d_in[2];
  const float* Wq   = (const float*)d_in[3];
  const float* Wk   = (const float*)d_in[4];
  const float* Wv   = (const float*)d_in[5];
  const float* Wg   = (const float*)d_in[6];
  const float* Wo   = (const float*)d_in[7];
  const float* memk = (const float*)d_in[8];
  const float* memv = (const float*)d_in[9];
  const float* ffg  = (const float*)d_in[10];
  const float* ffb  = (const float*)d_in[11];
  const float* W1   = (const float*)d_in[12];
  const float* b1   = (const float*)d_in[13];
  const float* W2   = (const float*)d_in[14];
  const float* b2   = (const float*)d_in[15];

  char* w = (char*)d_ws; size_t off = 0;
  auto alloc = [&](size_t bytes) -> char* {
    char* p = w + off;
    off = (off + bytes + 255) & ~(size_t)255;
    return p;
  };

  float*  xbuf = (float*) alloc((size_t)TOK * DM * 4);
  bf16_t* hbuf = (bf16_t*)alloc((size_t)TOK * DM * 2);
  float*  qbuf = (float*) alloc((size_t)TOK * DM * 4);
  float*  kbuf = (float*) alloc((size_t)TOK * DM * 4);
  float*  vbuf = (float*) alloc((size_t)TOK * DM * 4);
  float*  kcb  = (float*) alloc((size_t)BAT * NCMP * DM * 4);
  float*  vcb  = (float*) alloc((size_t)BAT * NCMP * DM * 4);
  bf16_t* obuf = (bf16_t*)alloc((size_t)TOK * DM * 2);
  bf16_t* fbuf = (bf16_t*)alloc((size_t)TOK * DFF * 2);
  bf16_t* wbA  = (bf16_t*)alloc((size_t)DM * DFF * 2);
  bf16_t* wbB  = (bf16_t*)alloc((size_t)DM * DFF * 2);
  bf16_t* wbC  = (bf16_t*)alloc((size_t)DM * DM * 2);

  (void)hipMemcpyAsync(xbuf, x, (size_t)TOK * DM * 4, hipMemcpyDeviceToDevice, stream);

  auto cvt = [&](const float* s, bf16_t* d, int n) {
    cvt_f32_bf16<<<(n + 255) / 256, 256, 0, stream>>>(s, d, n);
  };
  auto gemm = [&](const bf16_t* A, const bf16_t* Bm, float* Cf, bf16_t* Cb,
                  const float* bias, const float* resid, int M, int N, int K, int mode) {
    dim3 g(N / BN, M / BM);
    gemm_wmma<<<g, 256, 0, stream>>>(A, Bm, Cf, Cb, bias, resid, M, N, K, mode);
  };

  for (int l = 0; l < 2; l++) {
    for (int a = 0; a < 2; a++) {
      int la = l * 2 + a;
      ln_bf16<<<TOK / 8, 256, 0, stream>>>(xbuf, lng + (size_t)la * DM, lnb + (size_t)la * DM, hbuf, TOK);
      cvt(Wq + (size_t)la * DM * DM, wbA, DM * DM);
      cvt(Wk + (size_t)la * DM * DM, wbB, DM * DM);
      cvt(Wv + (size_t)la * DM * DM, wbC, DM * DM);
      gemm(hbuf, wbA, qbuf, nullptr, nullptr, nullptr, TOK, DM, DM, 0);
      gemm(hbuf, wbB, kbuf, nullptr, nullptr, nullptr, TOK, DM, DM, 0);
      gemm(hbuf, wbC, vbuf, nullptr, nullptr, nullptr, TOK, DM, DM, 0);
      int tot = BAT * NCMP * DM;
      compress_kv<<<(tot + 255) / 256, 256, 0, stream>>>(kbuf, memk + (size_t)la * NH * HD, kcb);
      compress_kv<<<(tot + 255) / 256, 256, 0, stream>>>(vbuf, memv + (size_t)la * NH * HD, vcb);
      nsa_attn<<<TOK, 256, 0, stream>>>(qbuf, kbuf, vbuf, kcb, vcb, hbuf,
                                        Wg + (size_t)la * DM * 3 * NH, obuf);
      cvt(Wo + (size_t)la * DM * DM, wbA, DM * DM);
      gemm(obuf, wbA, xbuf, nullptr, nullptr, xbuf, TOK, DM, DM, 1);
    }
    // feed-forward
    ln_bf16<<<TOK / 8, 256, 0, stream>>>(xbuf, ffg + (size_t)l * DM, ffb + (size_t)l * DM, hbuf, TOK);
    cvt(W1 + (size_t)l * DM * DFF, wbA, DM * DFF);
    gemm(hbuf, wbA, nullptr, fbuf, b1 + (size_t)l * DFF, nullptr, TOK, DFF, DM, 2);
    cvt(W2 + (size_t)l * DFF * DM, wbB, DFF * DM);
    gemm(fbuf, wbB, xbuf, nullptr, b2 + (size_t)l * DM, xbuf, TOK, DM, DFF, 1);
  }

  (void)hipMemcpyAsync(d_out, xbuf, (size_t)TOK * DM * 4, hipMemcpyDeviceToDevice, stream);
}
